// Attention_2705829396770
// MI455X (gfx1250) — matrix operand order repaired and checked by reading, not compiled, not run on hardware
//
#include <hip/hip_runtime.h>
#include <math.h>

// ---------------------------------------------------------------- constants
#define BB     2
#define NN     4096
#define CC     768
#define HH     12
#define DD     64
#define CPBH   512
#define MM     (BB * NN)          // 8192 rows for both GEMMs

typedef __bf16 bf16_t;
typedef bf16_t v16bf __attribute__((ext_vector_type(16)));
typedef float  v8f   __attribute__((ext_vector_type(8)));
typedef unsigned v4u __attribute__((ext_vector_type(4)));

union FragBF { v4u u[2]; v16bf f; };

// pack two fp32 -> packed bf16 pair (round to nearest even-ish)
__device__ inline unsigned pk_bf16(float a, float b) {
    unsigned ua = __float_as_uint(a);
    unsigned ub = __float_as_uint(b);
    ua = (ua + 0x7FFFu + ((ua >> 16) & 1u)) >> 16;
    ub = (ub + 0x7FFFu + ((ub >> 16) & 1u)) >> 16;
    return (ua & 0xFFFFu) | (ub << 16);
}

// order-preserving float -> uint encoding for atomicMax
__device__ inline unsigned enc_f(float f) {
    unsigned u = __float_as_uint(f);
    return (u & 0x80000000u) ? ~u : (u | 0x80000000u);
}
__device__ inline float dec_f(unsigned u) {
    u = (u & 0x80000000u) ? (u & 0x7FFFFFFFu) : ~u;
    return __uint_as_float(u);
}

// ---------------------------------------------------------------- bf16 WMMA GEMM
// C[M x Nn] = A[M x K] * B[K x Nn] + bias[Nn]; M%64==0, Nn%128==0, K%32==0.
// 256 threads = 8 waves (2x4); WG tile 64x128; wave tile 32x32 (4 C-frags).
// Double-buffered LDS + register prefetch: next tile's global loads are issued
// before the WMMAs so HBM latency overlaps the matrix pipe.
template <int K, int Nn>
__global__ __launch_bounds__(256) void gemm_bf16_wmma(
    const float* __restrict__ A, const float* __restrict__ Bm,
    float* __restrict__ C, const float* __restrict__ bias, int M)
{
    __shared__ __attribute__((aligned(16))) unsigned short sA[2][64 * 32];   // [m][k] bf16
    __shared__ __attribute__((aligned(16))) unsigned short sB[2][128 * 32];  // [n][k] bf16 (B^T)

    const int tid  = threadIdx.x;
    const int lane = tid & 31;
    const int wave = tid >> 5;
    const int wm   = wave >> 2;                  // 0..1
    const int wn   = wave & 3;                   // 0..3
    const int m0   = blockIdx.y * 64;
    const int n0   = blockIdx.x * 128;

    float4 aReg[2];
    float  bReg[2][8];

    auto loadRegs = [&](int k0) {
        for (int r = 0; r < 2; ++r) {
            int idx4 = tid + r * 256;            // 512 float4 in 64x32 tile
            int m = idx4 >> 3, kq = idx4 & 7;
            aReg[r] = *(const float4*)(A + (size_t)(m0 + m) * K + k0 + kq * 4);
        }
        for (int r = 0; r < 2; ++r) {
            int idx = tid + r * 256;             // 512 (n,kq) units
            int n = idx & 127, kq = idx >> 7;    // kq: 0..3, k = kq*8..+8
            for (int i = 0; i < 8; ++i)
                bReg[r][i] = Bm[(size_t)(k0 + kq * 8 + i) * Nn + n0 + n];
        }
    };
    auto packStore = [&](int buf) {
        for (int r = 0; r < 2; ++r) {
            int idx4 = tid + r * 256;
            int m = idx4 >> 3, kq = idx4 & 7;
            unsigned* dst = (unsigned*)sA[buf] + m * 16 + kq * 2;
            dst[0] = pk_bf16(aReg[r].x, aReg[r].y);
            dst[1] = pk_bf16(aReg[r].z, aReg[r].w);
        }
        for (int r = 0; r < 2; ++r) {
            int idx = tid + r * 256;
            int n = idx & 127, kq = idx >> 7;
            v4u pv;
            pv[0] = pk_bf16(bReg[r][0], bReg[r][1]);
            pv[1] = pk_bf16(bReg[r][2], bReg[r][3]);
            pv[2] = pk_bf16(bReg[r][4], bReg[r][5]);
            pv[3] = pk_bf16(bReg[r][6], bReg[r][7]);
            *(v4u*)&sB[buf][n * 32 + kq * 8] = pv;
        }
    };

    v8f acc[2][2];
    for (int i = 0; i < 2; ++i)
        for (int j = 0; j < 2; ++j)
            for (int e = 0; e < 8; ++e) acc[i][j][e] = 0.0f;

    const int l15 = lane & 15;
    const int akb = (lane < 16) ? 0 : 8;         // A: K pairs 0..7/16..23 vs 8..15/24..31
    const int bkb = (lane < 16) ? 0 : 8;         // B: K 0..7/16..23 vs 8..15/24..31

    loadRegs(0);
    packStore(0);

    const int nk = K / 32;
    for (int kt = 0; kt < nk; ++kt) {
        __syncthreads();
        if (kt + 1 < nk) {
            loadRegs((kt + 1) * 32);             // in flight across the WMMAs
            // speculative prefetch two tiles ahead (L2-resident thereafter)
            if (kt + 2 < nk) {
                __builtin_prefetch(A + (size_t)(m0 + (tid >> 3)) * K + (kt + 2) * 32, 0, 1);
                __builtin_prefetch(Bm + (size_t)((kt + 2) * 32 + (tid >> 7) * 8) * Nn
                                      + n0 + (tid & 127), 0, 1);
            }
        }
        const int buf = kt & 1;
        FragBF afr[2], bfr[2];
        for (int ms = 0; ms < 2; ++ms) {
            const v4u* pa = (const v4u*)&sA[buf][(wm * 32 + ms * 16 + l15) * 32 + akb];
            afr[ms].u[0] = pa[0];
            afr[ms].u[1] = pa[2];
        }
        for (int ns = 0; ns < 2; ++ns) {
            const v4u* pb = (const v4u*)&sB[buf][(wn * 32 + ns * 16 + l15) * 32 + bkb];
            bfr[ns].u[0] = pb[0];
            bfr[ns].u[1] = pb[2];
        }
        for (int ms = 0; ms < 2; ++ms)
            for (int ns = 0; ns < 2; ++ns)
                acc[ms][ns] = __builtin_amdgcn_wmma_f32_16x16x32_bf16(
                    false, afr[ms].f, false, bfr[ns].f,
                    (short)0, acc[ms][ns], false, false);
        if (kt + 1 < nk) packStore((kt + 1) & 1);
    }

    // ---- store C (16x16 f32 C/D layout: lane&15 = col, lane>=16 -> rows +8)
    const int moff = (lane >= 16) ? 8 : 0;
    for (int ms = 0; ms < 2; ++ms)
        for (int ns = 0; ns < 2; ++ns) {
            int col = n0 + wn * 32 + ns * 16 + l15;
            float bv = bias[col];                // always valid (zero vector for QKV)
            for (int v = 0; v < 8; ++v) {
                int row = m0 + wm * 32 + ms * 16 + v + moff;
                C[(size_t)row * Nn + col] = acc[ms][ns][v] + bv;
            }
        }
}

// ---------------------------------------------------------------- CPB bias MLP (per edge, batch independent)
__global__ void cpb_kernel(const float* __restrict__ coords,
                           const int* __restrict__ e0, const int* __restrict__ e1,
                           const float* __restrict__ w1, const float* __restrict__ b1,
                           const float* __restrict__ w2, float* __restrict__ biasb, int E)
{
    int e = blockIdx.x * blockDim.x + threadIdx.x;
    if (e >= E) return;
    int a = e0[e], b = e1[e];
    float f0 = coords[a * 2], f1 = coords[a * 2 + 1];
    float f2 = coords[b * 2], f3 = coords[b * 2 + 1];
    float acc[HH];
    for (int j = 0; j < HH; ++j) acc[j] = 0.0f;
    for (int k = 0; k < CPBH; ++k) {
        float t = f0 * w1[k] + f1 * w1[CPBH + k] + f2 * w1[2 * CPBH + k]
                + f3 * w1[3 * CPBH + k] + b1[k];
        float g = 0.5f * t * (1.0f + erff(t * 0.70710678118654752f));  // exact GELU
        const float* w2r = w2 + k * HH;
        for (int j = 0; j < HH; ++j) acc[j] += g * w2r[j];
    }
    for (int j = 0; j < HH; ++j)
        biasb[(size_t)e * HH + j] = 16.0f / (1.0f + expf(-acc[j]));
}

// ---------------------------------------------------------------- l2-normalize q,k in place
__global__ void normalize_kernel(float* __restrict__ qkv)
{
    int t = blockIdx.x * blockDim.x + threadIdx.x;
    const int tot = BB * NN * 2 * HH;
    if (t >= tot) return;
    int h = t % HH;
    int p = (t / HH) & 1;                 // 0=q, 1=k
    int n = (t / (HH * 2)) % NN;
    int b = t / (HH * 2 * NN);
    float* ptr = qkv + ((size_t)(b * NN + n) * 3 + p) * CC + h * DD;
    float s = 0.0f;
    const float4* p4 = (const float4*)ptr;
    for (int i = 0; i < DD / 4; ++i) {
        float4 v = p4[i];
        s += v.x * v.x + v.y * v.y + v.z * v.z + v.w * v.w;
    }
    float inv = 1.0f / fmaxf(sqrtf(s), 1e-12f);
    float4* o4 = (float4*)ptr;
    for (int i = 0; i < DD / 4; ++i) {
        float4 v = o4[i];
        v.x *= inv; v.y *= inv; v.z *= inv; v.w *= inv;
        o4[i] = v;
    }
}

// ---------------------------------------------------------------- init gmax + zero bias vector
__global__ void init_kernel(unsigned* __restrict__ gmax, float* __restrict__ zbias)
{
    int t = blockIdx.x * blockDim.x + threadIdx.x;
    if (t < BB * HH) gmax[t] = 0u;        // encoded -inf
    if (t < 3 * CC)  zbias[t] = 0.0f;
}

// ---------------------------------------------------------------- per-query edge segment bounds
__global__ void row_bounds_kernel(const int* __restrict__ e0,
                                  int* __restrict__ rs, int* __restrict__ re, int E)
{
    int i = blockIdx.x * blockDim.x + threadIdx.x;
    if (i >= E) return;
    int q = e0[i];
    if (i == 0     || e0[i - 1] != q) rs[q] = i;
    if (i == E - 1 || e0[i + 1] != q) re[q] = i + 1;
}

// ---------------------------------------------------------------- edge scores + global (b,h) max
__global__ void edge_score_kernel(const float* __restrict__ qkv,
                                  const float* __restrict__ biasb,
                                  const int* __restrict__ e0, const int* __restrict__ e1,
                                  const float* __restrict__ lls,
                                  float* __restrict__ sbuf, unsigned* __restrict__ gmax, int E)
{
    int t = blockIdx.x * blockDim.x + threadIdx.x;
    const long tot = (long)BB * E * HH;
    if (t >= tot) return;
    int h = t % HH;
    int e = (t / HH) % E;
    int b = t / (HH * E);
    const float4* q4 = (const float4*)(qkv + ((size_t)(b * NN + e0[e]) * 3 + 0) * CC + h * DD);
    const float4* k4 = (const float4*)(qkv + ((size_t)(b * NN + e1[e]) * 3 + 1) * CC + h * DD);
    float dot = 0.0f;
    for (int i = 0; i < DD / 4; ++i) {
        float4 qv = q4[i], kv = k4[i];
        dot += qv.x * kv.x + qv.y * kv.y + qv.z * kv.z + qv.w * kv.w;
    }
    float ls = expf(fminf(lls[h], 4.6051701859880914f));   // ln(100)
    float s  = ls * dot + biasb[(size_t)e * HH + h];
    sbuf[t] = s;
    atomicMax(&gmax[b * HH + h], enc_f(s));
}

// ---------------------------------------------------------------- softmax + weighted V scatter, one wave per (b,n)
__global__ __launch_bounds__(256) void attn_out_kernel(
    const float* __restrict__ qkv, const float* __restrict__ sbuf,
    const unsigned* __restrict__ gmax,
    const int* __restrict__ e1, const int* __restrict__ rs, const int* __restrict__ re,
    float* __restrict__ attn, int E)
{
    int wid  = blockIdx.x * 8 + (threadIdx.x >> 5);
    int lane = threadIdx.x & 31;
    if (wid >= BB * NN) return;
    int b = wid / NN;
    int n = wid % NN;
    int r0 = rs[n], r1 = re[n];

    float gm[HH], denom[HH];
    for (int h = 0; h < HH; ++h) { gm[h] = dec_f(gmax[b * HH + h]); denom[h] = 0.0f; }

    const float* srow = sbuf + (size_t)b * E * HH;
    for (int e = r0; e < r1; ++e)
        for (int h = 0; h < HH; ++h)
            denom[h] += expf(srow[(size_t)e * HH + h] - gm[h]);
    float rden[HH];
    for (int h = 0; h < HH; ++h) rden[h] = 1.0f / (denom[h] + 1e-6f);

    float acc[2 * HH];
    for (int i = 0; i < 2 * HH; ++i) acc[i] = 0.0f;

    for (int e = r0; e < r1; ++e) {
        int kv = e1[e];
        const float* vbase = qkv + ((size_t)(b * NN + kv) * 3 + 2) * CC;
        for (int h = 0; h < HH; ++h) {
            float w = expf(srow[(size_t)e * HH + h] - gm[h]) * rden[h];
            float2 vv = ((const float2*)(vbase + h * DD))[lane];   // dims 2*lane, 2*lane+1
            acc[2 * h]     += w * vv.x;
            acc[2 * h + 1] += w * vv.y;
        }
    }
    float* obase = attn + (size_t)(b * NN + n) * CC;
    for (int h = 0; h < HH; ++h) {
        float2 o; o.x = acc[2 * h]; o.y = acc[2 * h + 1];
        ((float2*)(obase + h * DD))[lane] = o;
    }
}

// ---------------------------------------------------------------- host launcher
extern "C" void kernel_launch(void* const* d_in, const int* in_sizes, int n_in,
                              void* d_out, int out_size, void* d_ws, size_t ws_size,
                              hipStream_t stream)
{
    const float* x      = (const float*)d_in[0];
    const float* w_qkv  = (const float*)d_in[1];
    const float* cpb_w1 = (const float*)d_in[2];
    const float* cpb_b1 = (const float*)d_in[3];
    const float* cpb_w2 = (const float*)d_in[4];
    const float* lls    = (const float*)d_in[5];
    const float* w_proj = (const float*)d_in[6];
    const float* b_proj = (const float*)d_in[7];
    const float* coords = (const float*)d_in[8];
    const int*   edges  = (const int*)d_in[9];
    float* out = (float*)d_out;

    const int E = in_sizes[9] / 2;
    const int* e0 = edges;
    const int* e1 = edges + E;

    // workspace layout
    float* qkv   = (float*)d_ws;                            // [B,N,3,H,D]
    float* sbuf  = qkv + (size_t)MM * 3 * CC;               // [B,E,H]
    float* biasb = sbuf + (size_t)BB * E * HH;              // [E,H]
    float* attn  = biasb + (size_t)E * HH;                  // [B,N,C]
    unsigned* gmax = (unsigned*)(attn + (size_t)MM * CC);   // [B,H]
    int* rs = (int*)(gmax + 32);
    int* re = rs + NN;
    float* zbias = (float*)(re + NN);                       // [3*C] zeros

    init_kernel<<<(3 * CC + 255) / 256, 256, 0, stream>>>(gmax, zbias);
    cpb_kernel<<<(E + 255) / 256, 256, 0, stream>>>(coords, e0, e1, cpb_w1, cpb_b1,
                                                    cpb_w2, biasb, E);
    row_bounds_kernel<<<(E + 255) / 256, 256, 0, stream>>>(e0, rs, re, E);

    // QKV GEMM: [8192,768] x [768,2304]
    gemm_bf16_wmma<CC, 3 * CC><<<dim3(3 * CC / 128, MM / 64), 256, 0, stream>>>(
        x, w_qkv, qkv, zbias, MM);

    normalize_kernel<<<(BB * NN * 2 * HH + 255) / 256, 256, 0, stream>>>(qkv);

    long tot = (long)BB * E * HH;
    edge_score_kernel<<<(int)((tot + 255) / 256), 256, 0, stream>>>(
        qkv, biasb, e0, e1, lls, sbuf, gmax, E);

    attn_out_kernel<<<(BB * NN) / 8, 256, 0, stream>>>(qkv, sbuf, gmax, e1, rs, re, attn, E);

    // Proj GEMM: [8192,768] x [768,768] + bias
    gemm_bf16_wmma<CC, CC><<<dim3(CC / 128, MM / 64), 256, 0, stream>>>(
        attn, w_proj, out, b_proj, MM);
}
